// LSTMVEDModule_3435973837098
// MI455X (gfx1250) — matrix operand order repaired and checked by reading, not compiled, not run on hardware
//
#include <hip/hip_runtime.h>

// ---------------------------------------------------------------------------
// LSTM VAE encoder-decoder for MI455X (gfx1250), wave32 + WMMA bf16.
// Persistent-kernel recurrence: weights live in LDS for all 1024 timesteps,
// timestep sync via grid-wide atomic barrier, gates computed with
// v_wmma_f32_16x16x32_bf16 and fused sigmoid/tanh cell update in registers.
// ---------------------------------------------------------------------------

typedef __attribute__((ext_vector_type(8)))  __bf16 v8bf;
typedef __attribute__((ext_vector_type(16))) __bf16 v16bf;
typedef __attribute__((ext_vector_type(8)))  float  v8f;

#define Bsz 128
#define Tsz 512
#define Fsz 64
#define Hsz 512
#define G4H 2048
#define KZ  576      // F + H (concatenated GEMM K)
#define NBLK 32
#define NTHR 256

union AF { v16bf v; v8bf h[2]; };

__device__ __forceinline__ v8f wmma_bf16(const AF& a, const AF& b, v8f c) {
  return __builtin_amdgcn_wmma_f32_16x16x32_bf16(false, a.v, false, b.v,
                                                 (short)0, c, false, false);
}

__device__ __forceinline__ float sigf(float x) { return 1.0f / (1.0f + __expf(-x)); }

// ---- grid-wide barrier (generation counter; works for any initial gen) ----
__device__ __forceinline__ void grid_sync(unsigned* cnt, unsigned* gen) {
  __syncthreads();
  if (threadIdx.x == 0) {
    __threadfence();
    unsigned g = __hip_atomic_load(gen, __ATOMIC_ACQUIRE, __HIP_MEMORY_SCOPE_AGENT);
    unsigned a = __hip_atomic_fetch_add(cnt, 1u, __ATOMIC_ACQ_REL, __HIP_MEMORY_SCOPE_AGENT);
    if (a == (unsigned)(NBLK - 1)) {
      __hip_atomic_store(cnt, 0u, __ATOMIC_RELAXED, __HIP_MEMORY_SCOPE_AGENT);
      __hip_atomic_fetch_add(gen, 1u, __ATOMIC_ACQ_REL, __HIP_MEMORY_SCOPE_AGENT);
    } else {
      while (__hip_atomic_load(gen, __ATOMIC_ACQUIRE, __HIP_MEMORY_SCOPE_AGENT) == g)
        __builtin_amdgcn_s_sleep(1);
    }
  }
  __syncthreads();
}

// ---- one LSTM cell step for this wave's 16x16 (batch, j) slab ----
// cur: z buffer [128][576] bf16 being read ([x | h])
// lW : LDS weight tile [64][576] (4 gates x 16 j-cols, full K)
// nxt: z buffer whose h-section receives h_new
__device__ __forceinline__ void lstm_gates(const __bf16* __restrict__ cur,
                                           const __bf16* __restrict__ lW,
                                           const float*  __restrict__ bias,
                                           float*        __restrict__ cst,
                                           __bf16*       __restrict__ nxt,
                                           int g, int m_base, int lr, int hi) {
  const __bf16* zr = cur + (m_base + lr) * KZ;
  const __bf16* w0 = lW + (0 * 16 + lr) * KZ;
  const __bf16* w1 = lW + (1 * 16 + lr) * KZ;
  const __bf16* w2 = lW + (2 * 16 + lr) * KZ;
  const __bf16* w3 = lW + (3 * 16 + lr) * KZ;
  v8f acc0 = {}, acc1 = {}, acc2 = {}, acc3 = {};
  for (int kk = 0; kk < KZ; kk += 32) {
    AF a, b0, b1, b2, b3;
    a.h[0] = *(const v8bf*)(zr + kk + hi * 8);
    a.h[1] = *(const v8bf*)(zr + kk + 16 + hi * 8);
    const int ko = kk + hi * 8;
    b0.h[0] = *(const v8bf*)(w0 + ko); b0.h[1] = *(const v8bf*)(w0 + ko + 16);
    b1.h[0] = *(const v8bf*)(w1 + ko); b1.h[1] = *(const v8bf*)(w1 + ko + 16);
    b2.h[0] = *(const v8bf*)(w2 + ko); b2.h[1] = *(const v8bf*)(w2 + ko + 16);
    b3.h[0] = *(const v8bf*)(w3 + ko); b3.h[1] = *(const v8bf*)(w3 + ko + 16);
    acc0 = wmma_bf16(a, b0, acc0);
    acc1 = wmma_bf16(a, b1, acc1);
    acc2 = wmma_bf16(a, b2, acc2);
    acc3 = wmma_bf16(a, b3, acc3);
  }
  const int j = g * 16 + lr;
  const float bi = bias[0 * Hsz + j];
  const float bf = bias[1 * Hsz + j];
  const float bg = bias[2 * Hsz + j];
  const float bo = bias[3 * Hsz + j];
#pragma unroll
  for (int i = 0; i < 8; ++i) {
    const int row = m_base + hi * 8 + i;
    const float ig = sigf(acc0[i] + bi);
    const float fg = sigf(acc1[i] + bf);
    const float gg = tanhf(acc2[i] + bg);
    const float og = sigf(acc3[i] + bo);
    const float cn = fg * cst[row * Hsz + j] + ig * gg;
    cst[row * Hsz + j] = cn;
    nxt[row * KZ + Fsz + j] = (__bf16)(og * tanhf(cn));
  }
}

// ---- K=512 projection tile (h @ W.T), arow points at h-section row ----
__device__ __forceinline__ v8f proj512(const __bf16* __restrict__ arow,
                                       const __bf16* __restrict__ brow, int hi) {
  v8f acc = {};
  for (int kk = 0; kk < Hsz; kk += 32) {
    AF a, b;
    a.h[0] = *(const v8bf*)(arow + kk + hi * 8);
    a.h[1] = *(const v8bf*)(arow + kk + 16 + hi * 8);
    const int ko = kk + hi * 8;
    b.h[0] = *(const v8bf*)(brow + ko);
    b.h[1] = *(const v8bf*)(brow + ko + 16);
    acc = wmma_bf16(a, b, acc);
  }
  return acc;
}

// ---------------------------------------------------------------------------
// Init: zero c-state, z1; z0 gets x_0 in its x-section; reset barrier.
// ---------------------------------------------------------------------------
__global__ void lstm_ved_init(const float* __restrict__ ts,
                              __bf16* __restrict__ z0, __bf16* __restrict__ z1,
                              float* __restrict__ cst, unsigned* __restrict__ bar) {
  const size_t i = (size_t)blockIdx.x * blockDim.x + threadIdx.x;
  const size_t stride = (size_t)gridDim.x * blockDim.x;
  for (size_t k = i; k < (size_t)Bsz * KZ; k += stride) {
    const int b = (int)(k / KZ), col = (int)(k % KZ);
    z0[k] = (col < Fsz) ? (__bf16)ts[(b * Tsz) * Fsz + col] : (__bf16)0.0f;
    z1[k] = (__bf16)0.0f;
  }
  for (size_t k = i; k < (size_t)Bsz * Hsz; k += stride) cst[k] = 0.0f;
  if (i < 2) bar[i] = 0u;
}

// ---------------------------------------------------------------------------
// Convert fp32 weights -> bf16 concatenated layouts in workspace.
// wce/wcd: [2048][576] = [W_ih | W_hh] per gate-row. wm:[512][512] wo:[64][512]
// ---------------------------------------------------------------------------
__global__ void lstm_ved_convert(const float* __restrict__ We_ih, const float* __restrict__ We_hh,
                                 const float* __restrict__ Wd_ih, const float* __restrict__ Wd_hh,
                                 const float* __restrict__ Wm,    const float* __restrict__ Wo,
                                 __bf16* __restrict__ wce, __bf16* __restrict__ wcd,
                                 __bf16* __restrict__ wmb, __bf16* __restrict__ wob) {
  const size_t i = (size_t)blockIdx.x * blockDim.x + threadIdx.x;
  const size_t stride = (size_t)gridDim.x * blockDim.x;
  for (size_t k = i; k < (size_t)G4H * KZ; k += stride) {
    const int n = (int)(k / KZ), col = (int)(k % KZ);
    wce[k] = (__bf16)((col < Fsz) ? We_ih[n * Fsz + col] : We_hh[n * Hsz + (col - Fsz)]);
    wcd[k] = (__bf16)((col < Fsz) ? Wd_ih[n * Fsz + col] : Wd_hh[n * Hsz + (col - Fsz)]);
  }
  for (size_t k = i; k < (size_t)Hsz * Hsz; k += stride) wmb[k] = (__bf16)Wm[k];
  for (size_t k = i; k < (size_t)Fsz * Hsz; k += stride) wob[k] = (__bf16)Wo[k];
}

// ---------------------------------------------------------------------------
// Persistent main kernel: 32 blocks x 256 threads (8 waves).
// Block g owns hidden-column tile j in [g*16, g*16+16); wave w owns batch rows
// [w*16, w*16+16). Encoder: 512 steps, 1 grid sync each (double-buffered z).
// Latent projection. Decoder: 512 steps, 2 grid syncs each (out-projection by
// blocks 0..3 feeds the cell input).
// ---------------------------------------------------------------------------
__global__ __launch_bounds__(NTHR, 1)
void lstm_ved_main(const float* __restrict__ ts,
                   const float* __restrict__ be, const float* __restrict__ bd,
                   const float* __restrict__ bm, const float* __restrict__ bo,
                   const __bf16* __restrict__ wce, const __bf16* __restrict__ wcd,
                   const __bf16* __restrict__ wmb, const __bf16* __restrict__ wob,
                   __bf16* __restrict__ z0, __bf16* __restrict__ z1,
                   float* __restrict__ cst, float* __restrict__ out,
                   unsigned* __restrict__ bar) {
  extern __shared__ __align__(16) char smem_raw[];
  __bf16* lWe = (__bf16*)smem_raw;        // 64 x 576 encoder gate tile (72KB)
  __bf16* lWd = lWe + 64 * KZ;            // 64 x 576 decoder gate tile (72KB)
  __bf16* lWo = lWd + 64 * KZ;            // 16 x 512 Wo tile (blocks 0..3, 16KB)

  const int tid = threadIdx.x;
  const int g = blockIdx.x;
  const int wave = tid >> 5;
  const int lane = tid & 31;
  const int lr = lane & 15;
  const int hi = lane >> 4;
  const int m_base = wave * 16;
  unsigned* cnt = bar;
  unsigned* gen = bar + 1;

  // ---- one-time LDS fill: weight tiles stay resident for all 1024 steps ----
  for (int idx = tid; idx < 64 * KZ; idx += NTHR) {
    const int r = idx / KZ, col = idx - r * KZ;
    const int gate = r >> 4, rr = r & 15;
    const int grow = gate * Hsz + g * 16 + rr;
    lWe[idx] = wce[grow * KZ + col];
    lWd[idx] = wcd[grow * KZ + col];
  }
  if (g < 4) {
    for (int idx = tid; idx < 16 * Hsz; idx += NTHR) {
      const int r = idx >> 9, col = idx & 511;
      lWo[idx] = wob[(g * 16 + r) * Hsz + col];
    }
  }
  __syncthreads();

  // ======================= encoder: 512 recurrent steps ====================
  for (int t = 0; t < Tsz; ++t) {
    const __bf16* cur = (t & 1) ? z1 : z0;
    __bf16* nxt = (t & 1) ? z0 : z1;
    lstm_gates(cur, lWe, be, cst, nxt, g, m_base, lr, hi);
    if (t + 1 < Tsz) {  // stage x_{t+1} into next z (8192 elems == grid size)
      const int idx = g * NTHR + tid;
      const int b = idx >> 6, f = idx & 63;
      nxt[b * KZ + f] = (__bf16)ts[(b * Tsz + t + 1) * Fsz + f];
    }
    grid_sync(cnt, gen);
  }

  // ================ latent mean: h_lat = h_enc @ Wm.T + bm =================
  {
    const __bf16* cur = z0;  // h_enc lives in z0 after 512 steps
    __bf16* nxt = z1;
    v8f acc = proj512(cur + (m_base + lr) * KZ + Fsz, wmb + (g * 16 + lr) * Hsz, hi);
    const int j = g * 16 + lr;
    const float bb = bm[j];
#pragma unroll
    for (int i = 0; i < 8; ++i) {
      const int row = m_base + hi * 8 + i;
      nxt[row * KZ + Fsz + j] = (__bf16)(acc[i] + bb);
    }
    grid_sync(cnt, gen);
  }

  // ======================= decoder: 512 recurrent steps ====================
  for (int i = 0; i < Tsz; ++i) {
    __bf16* cur = ((i + 1) & 1) ? z1 : z0;   // i=0 -> z1 (holds h_lat)
    __bf16* nxt = ((i + 1) & 1) ? z0 : z1;
    // phase 1: out = h @ Wo.T + bo  (blocks 0..3 cover f in 0..63)
    if (g < 4) {
      v8f acc = proj512(cur + (m_base + lr) * KZ + Fsz, lWo + lr * Hsz, hi);
      const int f = g * 16 + lr;
      const float bb = bo[f];
#pragma unroll
      for (int e = 0; e < 8; ++e) {
        const int row = m_base + hi * 8 + e;
        const float v = acc[e] + bb;
        out[(row * Tsz + (Tsz - 1 - i)) * Fsz + f] = v;  // chronological flip
        cur[row * KZ + f] = (__bf16)v;                   // feed back as x
      }
    }
    grid_sync(cnt, gen);
    // phase 2: LSTM cell on [out | h]
    lstm_gates(cur, lWd, bd, cst, nxt, g, m_base, lr, hi);
    grid_sync(cnt, gen);
  }
}

// ---------------------------------------------------------------------------
extern "C" void kernel_launch(void* const* d_in, const int* in_sizes, int n_in,
                              void* d_out, int out_size, void* d_ws, size_t ws_size,
                              hipStream_t stream) {
  (void)in_sizes; (void)n_in; (void)out_size; (void)ws_size;
  const float* ts    = (const float*)d_in[0];
  const float* We_ih = (const float*)d_in[1];
  const float* We_hh = (const float*)d_in[2];
  const float* be    = (const float*)d_in[3];
  const float* Wd_ih = (const float*)d_in[4];
  const float* Wd_hh = (const float*)d_in[5];
  const float* bd    = (const float*)d_in[6];
  const float* Wm    = (const float*)d_in[7];
  const float* bm    = (const float*)d_in[8];
  const float* Wo    = (const float*)d_in[9];
  const float* bo    = (const float*)d_in[10];

  // workspace layout (bytes, all 256-aligned); total ~5.87 MB
  char* ws = (char*)d_ws;
  __bf16*   wce = (__bf16*)(ws + 0);        // 2048*576 bf16 = 2359296 B
  __bf16*   wcd = (__bf16*)(ws + 2359296);  // 2048*576 bf16
  __bf16*   wmb = (__bf16*)(ws + 4718592);  // 512*512 bf16 = 524288 B
  __bf16*   wob = (__bf16*)(ws + 5242880);  // 64*512 bf16 = 65536 B
  __bf16*   z0  = (__bf16*)(ws + 5308416);  // 128*576 bf16 = 147456 B
  __bf16*   z1  = (__bf16*)(ws + 5455872);  // 128*576 bf16
  float*    cst = (float*) (ws + 5603328);  // 128*512 f32 = 262144 B
  unsigned* bar = (unsigned*)(ws + 5865472);

  lstm_ved_init<<<256, 256, 0, stream>>>(ts, z0, z1, cst, bar);
  lstm_ved_convert<<<512, 256, 0, stream>>>(We_ih, We_hh, Wd_ih, Wd_hh, Wm, Wo,
                                            wce, wcd, wmb, wob);
  const size_t smem = (size_t)(64 * KZ * 2 + 16 * Hsz) * sizeof(__bf16);  // 163840 B
  lstm_ved_main<<<NBLK, NTHR, smem, stream>>>(ts, be, bd, bm, bo,
                                              wce, wcd, wmb, wob,
                                              z0, z1, cst, (float*)d_out, bar);
}
